// OptimizedSentenceAwareEmbeddingModule_39256001086083
// MI455X (gfx1250) — matrix operand order repaired and checked by reading, not compiled, not run on hardware
//
#include <hip/hip_runtime.h>

typedef __bf16 bf16;
typedef __attribute__((ext_vector_type(16))) __bf16 v16bf;
typedef __attribute__((ext_vector_type(8)))  __bf16 v8bf;
typedef __attribute__((ext_vector_type(8)))  float  v8f;

#define B_   2
#define S_   2048
#define E_   1024
#define H_   16
#define D_   64
#define E3_  3072
#define M_   (B_ * S_)   // 4096 rows

union V16u { v16bf v; v8bf h[2]; };

// ---- WMMA fragment loaders (wave32, 16x16x32 bf16, contiguous k = 16*(lane/16) + s in both A and B) ----

// A-matrix 16x32 (MxK): lane(0-15)=row m, k-base 0; lane(16-31)=row m, k-base 16.
// v0..v3 hold K = kb..kb+7 (pairs), v4..v7 hold K = kb+8..kb+15.
__device__ __forceinline__ v16bf load_a_frag(const bf16* __restrict__ base, int ld) {
  int lane = threadIdx.x & 31;
  int m  = lane & 15;
  int kb = (lane >> 4) * 16;
  const bf16* row = base + (size_t)m * ld;
  V16u u;
  u.h[0] = *(const v8bf*)(row + kb);
  u.h[1] = *(const v8bf*)(row + kb + 8);
  return u.v;
}

// B-matrix 32x16 (KxN) where B[k][n] = W[n][k] (W row-major, leading dim ld):
// lane(0-15): col n, K=0..15 ; lane(16-31): col n, K=16..31 → 16 contiguous bf16 of W's row.
__device__ __forceinline__ v16bf load_bT_frag(const bf16* __restrict__ base, int ld) {
  int lane = threadIdx.x & 31;
  int n  = lane & 15;
  int kb = (lane >> 4) * 16;
  return *(const v16bf*)(base + (size_t)n * ld + kb);
}

// ---------------- elementwise / setup kernels ----------------

__global__ void cvt_f32_bf16(const float* __restrict__ in, bf16* __restrict__ out, int n) {
  for (int i = blockIdx.x * blockDim.x + threadIdx.x; i < n; i += gridDim.x * blockDim.x)
    out[i] = (bf16)in[i];
}

// inclusive/exclusive cumsum of breaking tokens (token == 13) per batch row
__global__ void seg_scan(const long long* __restrict__ tok,
                         int* __restrict__ seg, int* __restrict__ segp) {
  if (threadIdx.x != 0) return;
  int b = blockIdx.x;
  int run = 0;
  for (int i = 0; i < S_; ++i) {
    int brk = (tok[b * S_ + i] == 13) ? 1 : 0;
    segp[b * S_ + i] = run;     // cumsum up to i-1
    run += brk;
    seg[b * S_ + i] = run;      // inclusive cumsum
  }
}

// Vt[b][h][dd][s] = V[b][s][h*64+dd]  (V = third chunk of qkv)
__global__ void build_vt(const bf16* __restrict__ qkv, bf16* __restrict__ vt) {
  const int total = B_ * H_ * D_ * S_;
  for (int i = blockIdx.x * blockDim.x + threadIdx.x; i < total; i += gridDim.x * blockDim.x) {
    int s  = i & (S_ - 1);
    int dd = (i >> 11) & (D_ - 1);
    int h  = (i >> 17) & (H_ - 1);
    int b  = i >> 21;
    vt[i] = qkv[((size_t)(b * S_ + s)) * E3_ + 2 * E_ + h * D_ + dd];
  }
}

// ---------------- GEMM: C[M,N] = A[M,K] @ W[N,K]^T + bias ----------------
// One wave computes a 32x64 C tile: 2 A-fragments share each of 4 B-fragments
// (8 WMMA per k-step on 6 fragment loads), cutting L2 fragment traffic ~40%
// vs a 16x64 tile. Data is L2-resident (192MB), so no LDS blocking needed.

__global__ __launch_bounds__(256)
void gemm_bf16_wmma(const bf16* __restrict__ A, const bf16* __restrict__ W,
                    const float* __restrict__ bias,
                    bf16* __restrict__ Cbf, float* __restrict__ Cf,
                    int M, int N, int K) {
  int wave = (blockIdx.x * blockDim.x + threadIdx.x) >> 5;
  int tilesN = N >> 6;
  int tm = wave / tilesN;                 // 32-row tiles
  int tn = wave - tm * tilesN;
  if (tm * 32 >= M) return;
  int m0 = tm * 32, n0 = tn * 64;
  int lane = threadIdx.x & 31;
  int nlo = lane & 15, hi = lane >> 4;

  v8f zero = {};
  v8f acc[2][4];
#pragma unroll
  for (int h2 = 0; h2 < 2; ++h2)
#pragma unroll
    for (int c = 0; c < 4; ++c) acc[h2][c] = zero;

  // per-lane row pointers for software prefetch of the next k-slab
  const bf16* arow_pf = A + (size_t)(m0 + nlo) * K;
  const bf16* wrow_pf = W + (size_t)(n0 + nlo) * K;

  for (int k0 = 0; k0 < K; k0 += 32) {
    if (k0 + 128 < K) {
      __builtin_prefetch(arow_pf + k0 + 128, 0, 1);   // global_prefetch_b8
      __builtin_prefetch(wrow_pf + k0 + 128, 0, 1);
    }
    v16bf a0 = load_a_frag(A + (size_t)m0 * K + k0, K);
    v16bf a1 = load_a_frag(A + (size_t)(m0 + 16) * K + k0, K);
#pragma unroll
    for (int c = 0; c < 4; ++c) {
      v16bf b = load_bT_frag(W + (size_t)(n0 + c * 16) * K + k0, K);
      acc[0][c] = __builtin_amdgcn_wmma_f32_16x16x32_bf16(
          false, a0, false, b, (short)0, acc[0][c], false, false);
      acc[1][c] = __builtin_amdgcn_wmma_f32_16x16x32_bf16(
          false, a1, false, b, (short)0, acc[1][c], false, false);
    }
  }

#pragma unroll
  for (int c = 0; c < 4; ++c) {
    int col = n0 + c * 16 + nlo;
    float bv = bias ? bias[col] : 0.0f;
#pragma unroll
    for (int h2 = 0; h2 < 2; ++h2) {
#pragma unroll
      for (int r = 0; r < 8; ++r) {
        int row = m0 + h2 * 16 + r + 8 * hi;   // C/D layout: lanes 0-15 → M=r, 16-31 → M=r+8
        float v = acc[h2][c][r] + bv;
        if (Cbf) Cbf[(size_t)row * N + col] = (bf16)v;
        else     Cf [(size_t)row * N + col] = v;
      }
    }
  }
}

// ---------------- flash attention with (inverted) sentence mask ----------------
// Reference blocks (k<=q && same-sentence); everything else is attended.
// One wave per (b, h, 16-query tile); streams all 2048 keys, 32 per iteration.

__global__ __launch_bounds__(32)
void flash_attn(const bf16* __restrict__ qkv, const bf16* __restrict__ vt,
                const int* __restrict__ seg, const int* __restrict__ segp,
                bf16* __restrict__ ctx) {
  __shared__ __align__(32) bf16 p_lds[16 * 32];

  int tile = blockIdx.x;                 // B_*H_*(S_/16)
  int qt = tile & (S_ / 16 - 1);
  int h  = (tile >> 7) & (H_ - 1);
  int b  = tile >> 11;
  int q0 = qt * 16;

  int lane = threadIdx.x & 31;
  int nlo = lane & 15, hi = lane >> 4;

  const bf16* qbase = qkv + (size_t)(b * S_) * E3_ + h * D_;
  const bf16* kbase = qkv + (size_t)(b * S_) * E3_ + E_ + h * D_;
  const bf16* vbase = vt + ((size_t)(b * H_ + h)) * D_ * S_;   // [dd][s]

  // Q A-fragments for the two 32-wide d chunks
  v16bf qa[2];
  {
    const bf16* qrow = qbase + (size_t)(q0 + nlo) * E3_;
#pragma unroll
    for (int c = 0; c < 2; ++c) {
      V16u u;
      u.h[0] = *(const v8bf*)(qrow + c * 32 + hi * 16);
      u.h[1] = *(const v8bf*)(qrow + c * 32 + hi * 16 + 8);
      qa[c] = u.v;
    }
  }

  int segq[8];
#pragma unroll
  for (int r = 0; r < 8; ++r) segq[r] = seg[b * S_ + q0 + r + 8 * hi];

  float mst[8], lst[8];
  v8f zero = {};
  v8f acc[4];
#pragma unroll
  for (int r = 0; r < 8; ++r) { mst[r] = -__builtin_inff(); lst[r] = 0.0f; }
#pragma unroll
  for (int c = 0; c < 4; ++c) acc[c] = zero;

  const float scale = 0.125f;            // 1/sqrt(64)

  for (int j = 0; j < S_; j += 32) {
    // ---- scores: two 16x16 tiles (keys j..j+15 and j+16..j+31), K = d = 64 ----
    v8f slo = zero, shi = zero;
    const bf16* krow_lo = kbase + (size_t)(j + nlo) * E3_;
    const bf16* krow_hi = kbase + (size_t)(j + 16 + nlo) * E3_;
#pragma unroll
    for (int c = 0; c < 2; ++c) {
      v16bf blo = *(const v16bf*)(krow_lo + c * 32 + hi * 16);
      v16bf bhi = *(const v16bf*)(krow_hi + c * 32 + hi * 16);
      slo = __builtin_amdgcn_wmma_f32_16x16x32_bf16(false, qa[c], false, blo, (short)0, slo, false, false);
      shi = __builtin_amdgcn_wmma_f32_16x16x32_bf16(false, qa[c], false, bhi, (short)0, shi, false, false);
    }

    int klo = j + nlo, khi = j + 16 + nlo;
    int sp_lo = segp[b * S_ + klo];
    int sp_hi = segp[b * S_ + khi];

    // ---- masked online softmax, per owned row r (M = r + 8*hi) ----
#pragma unroll
    for (int r = 0; r < 8; ++r) {
      int qi = q0 + r + 8 * hi;
      // reference semantics: BLOCK (k<=q && same sentence), allow everything else
      bool a0 = !((klo <= qi) && (segq[r] == sp_lo));
      bool a1 = !((khi <= qi) && (segq[r] == sp_hi));
      float v0 = a0 ? slo[r] * scale : -__builtin_inff();
      float v1 = a1 ? shi[r] * scale : -__builtin_inff();

      float rmax = fmaxf(v0, v1);
#pragma unroll
      for (int d = 1; d < 16; d <<= 1) rmax = fmaxf(rmax, __shfl_xor(rmax, d, 32));

      float mnew = fmaxf(mst[r], rmax);
      float sc = 1.0f, p0 = 0.0f, p1 = 0.0f;
      if (mnew != -__builtin_inff()) {
        sc = __expf(mst[r] - mnew);
        p0 = a0 ? __expf(v0 - mnew) : 0.0f;
        p1 = a1 ? __expf(v1 - mnew) : 0.0f;
      }
      float rs = p0 + p1;
#pragma unroll
      for (int d = 1; d < 16; d <<= 1) rs += __shfl_xor(rs, d, 32);

      lst[r] = lst[r] * sc + rs;
      mst[r] = mnew;
#pragma unroll
      for (int c = 0; c < 4; ++c) acc[c][r] *= sc;

      // stage P (16x32) through LDS for C-layout -> A-layout re-tiling
      p_lds[(r + 8 * hi) * 32 + nlo]      = (bf16)p0;
      p_lds[(r + 8 * hi) * 32 + 16 + nlo] = (bf16)p1;
    }

    asm volatile("s_wait_dscnt 0" ::: "memory");

    // reload P as an A fragment (16x32)
    V16u u;
    u.h[0] = *(const v8bf*)(&p_lds[nlo * 32 + (hi * 16)]);
    u.h[1] = *(const v8bf*)(&p_lds[nlo * 32 + (hi * 16) + 8]);
    v16bf pa = u.v;

    // ---- acc += P @ V  (4 d-chunks of 16, K=32 keys) ----
#pragma unroll
    for (int c = 0; c < 4; ++c) {
      const bf16* vptr = vbase + (size_t)(c * 16 + nlo) * S_ + j + hi * 16;
      v16bf vb = *(const v16bf*)vptr;
      acc[c] = __builtin_amdgcn_wmma_f32_16x16x32_bf16(false, pa, false, vb, (short)0, acc[c], false, false);
    }
  }

  // ---- normalize & write context (bf16 [B,S,E]) ----
  bf16* obase = ctx + (size_t)(b * S_) * E_ + h * D_;
#pragma unroll
  for (int c = 0; c < 4; ++c) {
    int col = c * 16 + nlo;
#pragma unroll
    for (int r = 0; r < 8; ++r) {
      int row = q0 + r + 8 * hi;
      float l = lst[r];
      float v = (l > 0.0f) ? acc[c][r] / l : 0.0f;
      obase[(size_t)row * E_ + col] = (bf16)v;
    }
  }
}

// ---------------- host launcher ----------------

extern "C" void kernel_launch(void* const* d_in, const int* in_sizes, int n_in,
                              void* d_out, int out_size, void* d_ws, size_t ws_size,
                              hipStream_t stream) {
  const float*     x   = (const float*)d_in[0];
  const long long* tok = (const long long*)d_in[1];
  const float*     wi  = (const float*)d_in[2];
  const float*     bi  = (const float*)d_in[3];
  const float*     wo  = (const float*)d_in[4];
  const float*     bo  = (const float*)d_in[5];
  float* out = (float*)d_out;

  char* ws = (char*)d_ws;
  size_t off = 0;
  auto alloc = [&](size_t bytes) -> void* {
    off = (off + 255) & ~(size_t)255;
    void* p = ws + off;
    off += bytes;
    return p;
  };
  bf16* xb   = (bf16*)alloc((size_t)M_ * E_ * 2);        //  8.4 MB
  bf16* wib  = (bf16*)alloc((size_t)E3_ * E_ * 2);       //  6.3 MB
  bf16* wob  = (bf16*)alloc((size_t)E_ * E_ * 2);        //  2.1 MB
  bf16* qkvb = (bf16*)alloc((size_t)M_ * E3_ * 2);       // 25.2 MB
  bf16* vtb  = (bf16*)alloc((size_t)B_ * H_ * D_ * S_ * 2); // 8.4 MB
  bf16* ctx  = (bf16*)alloc((size_t)M_ * E_ * 2);        //  8.4 MB
  int*  seg  = (int*)alloc((size_t)B_ * S_ * 4);
  int*  segp = (int*)alloc((size_t)B_ * S_ * 4);

  cvt_f32_bf16<<<1024, 256, 0, stream>>>(x,  xb,  M_ * E_);
  cvt_f32_bf16<<<1024, 256, 0, stream>>>(wi, wib, E3_ * E_);
  cvt_f32_bf16<<<512,  256, 0, stream>>>(wo, wob, E_ * E_);
  seg_scan<<<B_, 32, 0, stream>>>(tok, seg, segp);

  // QKV projection: [4096,1024] x [1024,3072] -> bf16 qkv (bias fused)
  {
    int waves = (M_ / 32) * (E3_ / 64);                  // 6144
    gemm_bf16_wmma<<<waves * 32 / 256, 256, 0, stream>>>(
        xb, wib, bi, qkvb, nullptr, M_, E3_, E_);
  }

  build_vt<<<2048, 256, 0, stream>>>(qkvb, vtb);

  flash_attn<<<B_ * H_ * (S_ / 16), 32, 0, stream>>>(qkvb, vtb, seg, segp, ctx);

  // Output projection: [4096,1024] x [1024,1024] -> f32 out (bias fused)
  {
    int waves = (M_ / 32) * (E_ / 64);                   // 2048
    gemm_bf16_wmma<<<waves * 32 / 256, 256, 0, stream>>>(
        ctx, wob, bo, nullptr, out, M_, E_, E_);
  }
}
